// GraphAttentionLayer_54485955117720
// MI455X (gfx1250) — matrix operand order repaired and checked by reading, not compiled, not run on hardware
//
#include <hip/hip_runtime.h>

typedef __attribute__((ext_vector_type(16))) _Float16 v16h;
typedef __attribute__((ext_vector_type(8)))  float    v8f;

#define N_NODES 8192
#define IN_DIM  256
#define OUT_DIM 64
#define GAT_ALPHA 0.2f
#define LOG2E 1.4426950408889634f
#define NSLICE 8
#define SLICE_W (N_NODES / NSLICE)   // 1024 columns per slice

// ---------------------------------------------------------------------------
// Phase 1: Wh = h @ W  via V_WMMA_F32_16X16X32_F16.
// Each wave computes a 16x64 tile of Wh. Writes f32 Wh (for f1/f2 precision)
// and an o-major f16 transpose WhT[o][node] so phase-2 B-operands are
// contiguous b128 loads.
// ---------------------------------------------------------------------------
__global__ void k_wh(const float* __restrict__ h, const float* __restrict__ W,
                     float* __restrict__ Whf, _Float16* __restrict__ WhT) {
  const int lane = threadIdx.x & 31;
  const int wave = (blockIdx.x * blockDim.x + threadIdx.x) >> 5;   // 0..511
  const int r0 = wave * 16;
  const int m = lane & 15;
  const int koffA = (lane < 16) ? 0 : 8;    // A-operand K group
  const int koffB = (lane < 16) ? 0 : 16;   // B-operand K group
  const int row = r0 + m;

  v8f d0 = {}, d1 = {}, d2 = {}, d3 = {};

  for (int kk = 0; kk < IN_DIM; kk += 32) {
    // A operand: h[row][kk + koffA + {0..7, 16..23}] -> f16
    const float* hp = h + (size_t)row * IN_DIM + kk + koffA;
    float4 ha = *(const float4*)(hp + 0);
    float4 hb = *(const float4*)(hp + 4);
    float4 hc = *(const float4*)(hp + 16);
    float4 hd = *(const float4*)(hp + 20);
    v16h av;
    av[0]=(_Float16)ha.x; av[1]=(_Float16)ha.y; av[2]=(_Float16)ha.z; av[3]=(_Float16)ha.w;
    av[4]=(_Float16)hb.x; av[5]=(_Float16)hb.y; av[6]=(_Float16)hb.z; av[7]=(_Float16)hb.w;
    av[8]=(_Float16)hc.x; av[9]=(_Float16)hc.y; av[10]=(_Float16)hc.z; av[11]=(_Float16)hc.w;
    av[12]=(_Float16)hd.x; av[13]=(_Float16)hd.y; av[14]=(_Float16)hd.z; av[15]=(_Float16)hd.w;

    // B operands: W[kk + koffA + {0..7, 16..23}][o0 + n], 4 o-tiles
    v16h b0, b1, b2, b3;
#pragma unroll
    for (int e = 0; e < 16; ++e) {
      const float* wr = W + (size_t)(kk + koffA + (e & 7) + ((e >> 3) << 4)) * OUT_DIM + m;
      b0[e] = (_Float16)wr[0];
      b1[e] = (_Float16)wr[16];
      b2[e] = (_Float16)wr[32];
      b3[e] = (_Float16)wr[48];
    }
    d0 = __builtin_amdgcn_wmma_f32_16x16x32_f16(false, av, false, b0, (short)0, d0, false, false);
    d1 = __builtin_amdgcn_wmma_f32_16x16x32_f16(false, av, false, b1, (short)0, d1, false, false);
    d2 = __builtin_amdgcn_wmma_f32_16x16x32_f16(false, av, false, b2, (short)0, d2, false, false);
    d3 = __builtin_amdgcn_wmma_f32_16x16x32_f16(false, av, false, b3, (short)0, d3, false, false);
  }

  // C/D layout: VGPR r, lanes 0-15 -> M=r, lanes 16-31 -> M=r+8; N = lane&15
  const int rhalf = (lane >> 4) * 8;
#pragma unroll
  for (int r = 0; r < 8; ++r) {
    int orow = r0 + r + rhalf;
    float v0 = d0[r], v1 = d1[r], v2 = d2[r], v3 = d3[r];
    float* wout = Whf + (size_t)orow * OUT_DIM + m;
    wout[0]  = v0;  wout[16] = v1;  wout[32] = v2;  wout[48] = v3;
    WhT[(size_t)(m +  0) * N_NODES + orow] = (_Float16)v0;
    WhT[(size_t)(m + 16) * N_NODES + orow] = (_Float16)v1;
    WhT[(size_t)(m + 32) * N_NODES + orow] = (_Float16)v2;
    WhT[(size_t)(m + 48) * N_NODES + orow] = (_Float16)v3;
  }
}

// ---------------------------------------------------------------------------
// Phase 1b: f1 = Wh @ a1, f2 = Wh @ a2  (one thread per node)
// ---------------------------------------------------------------------------
__global__ void k_f12(const float* __restrict__ Whf, const float* __restrict__ a,
                      float* __restrict__ f1, float* __restrict__ f2) {
  int i = blockIdx.x * blockDim.x + threadIdx.x;
  const float* wr = Whf + (size_t)i * OUT_DIM;
  float s1 = 0.f, s2 = 0.f;
#pragma unroll 8
  for (int o = 0; o < OUT_DIM; ++o) {
    float w = wr[o];
    s1 += w * a[o];
    s2 += w * a[OUT_DIM + o];
  }
  f1[i] = s1;
  f2[i] = s2;
}

// ---------------------------------------------------------------------------
// Phase 1c: global max of f2 (deterministic single-block reduction).
// leakyrelu(f1_i + max_j f2_j) >= every e_ij, so it is a safe softmax shift.
// ---------------------------------------------------------------------------
__global__ void k_f2max(const float* __restrict__ f2, float* __restrict__ f2m) {
  __shared__ float sm[256];
  float v = -3.0e38f;
  for (int i = threadIdx.x; i < N_NODES; i += 256) v = fmaxf(v, f2[i]);
  sm[threadIdx.x] = v;
  __syncthreads();
  for (int s = 128; s > 0; s >>= 1) {
    if ((int)threadIdx.x < s) sm[threadIdx.x] = fmaxf(sm[threadIdx.x], sm[threadIdx.x + s]);
    __syncthreads();
  }
  if (threadIdx.x == 0) *f2m = sm[0];
}

// ---------------------------------------------------------------------------
// Phase 2: single-pass masked-softmax attention (flash style, fixed shift).
// Wave = (16-row tile) x (1024-column slice). Builds P directly in the WMMA
// f16 A-operand layout; 4 WMMAs accumulate acc += P @ Wh and a 5th WMMA
// against an all-ones B operand accumulates the softmax denominators
// (row sums of P) in its D registers -- the XDL op co-executes with the
// surrounding VALU, removing 16 scalar adds/lane/chunk and all cross-lane
// reduction. Slices combine via f32 atomics.
// ---------------------------------------------------------------------------
__global__ void __launch_bounds__(256)
k_attn(const int* __restrict__ A, const _Float16* __restrict__ WhT,
       const float* __restrict__ f1, const float* __restrict__ f2,
       const float* __restrict__ f2m, float* __restrict__ acc,
       float* __restrict__ lbuf) {
  const int lane  = threadIdx.x & 31;
  const int wave  = (blockIdx.x * blockDim.x + threadIdx.x) >> 5;
  const int tile  = wave >> 3;          // 0..511 row tile
  const int slice = wave & (NSLICE - 1);
  const int r0 = tile * 16;
  const int c0 = slice * SLICE_W;
  const int m = lane & 15;
  const int koffA = (lane < 16) ? 0 : 16;
  const int koffB = (lane < 16) ? 0 : 16;
  const int row = r0 + m;

  const float f1v = f1[row];
  const float mx  = f1v + *f2m;
  const float rm  = fmaxf(mx, GAT_ALPHA * mx);   // leakyrelu upper bound for this row
  const float rmL = rm * LOG2E;                  // folded into one FMA per element

  v16h bones;
#pragma unroll
  for (int e = 0; e < 16; ++e) bones[e] = (_Float16)1.0f;

  v8f d0 = {}, d1 = {}, d2 = {}, d3 = {}, d4 = {};
  const int* Arow = A + (size_t)row * N_NODES;

  for (int j = c0; j < c0 + SLICE_W; j += 32) {
    __builtin_prefetch(Arow + j + 512, 0, 1);   // global_prefetch_b8, ~16 chunks ahead

    // ---- adjacency + f2 for this lane's 16 K columns ----
    int4  a0 = *(const int4*)(Arow + j + koffA + 0);
    int4  a1 = *(const int4*)(Arow + j + koffA + 4);
    int4  a2 = *(const int4*)(Arow + j + koffA + 8);
    int4  a3 = *(const int4*)(Arow + j + koffA + 12);
    float4 g0 = *(const float4*)(f2 + j + koffA + 0);
    float4 g1 = *(const float4*)(f2 + j + koffA + 4);
    float4 g2 = *(const float4*)(f2 + j + koffA + 8);
    float4 g3 = *(const float4*)(f2 + j + koffA + 12);

    v16h p;
    auto quad = [&](int ax, int ay, int az, int aw, float4 g, int eb) {
      float e, lr, pv;
      e = f1v + g.x; lr = fmaxf(e, GAT_ALPHA * e);
      pv = (ax > 0) ? __builtin_amdgcn_exp2f(__builtin_fmaf(lr, LOG2E, -rmL)) : 0.f;
      p[eb + 0] = (_Float16)pv;
      e = f1v + g.y; lr = fmaxf(e, GAT_ALPHA * e);
      pv = (ay > 0) ? __builtin_amdgcn_exp2f(__builtin_fmaf(lr, LOG2E, -rmL)) : 0.f;
      p[eb + 1] = (_Float16)pv;
      e = f1v + g.z; lr = fmaxf(e, GAT_ALPHA * e);
      pv = (az > 0) ? __builtin_amdgcn_exp2f(__builtin_fmaf(lr, LOG2E, -rmL)) : 0.f;
      p[eb + 2] = (_Float16)pv;
      e = f1v + g.w; lr = fmaxf(e, GAT_ALPHA * e);
      pv = (aw > 0) ? __builtin_amdgcn_exp2f(__builtin_fmaf(lr, LOG2E, -rmL)) : 0.f;
      p[eb + 3] = (_Float16)pv;
    };
    quad(a0.x, a0.y, a0.z, a0.w, g0, 0);
    quad(a1.x, a1.y, a1.z, a1.w, g1, 4);
    quad(a2.x, a2.y, a2.z, a2.w, g2, 8);
    quad(a3.x, a3.y, a3.z, a3.w, g3, 12);

    // ---- B operands: WhT[o0+n][j + koffB + 0..15], contiguous f16 ----
    const _Float16* bp = WhT + j + koffB + (size_t)m * N_NODES;
    v16h b0 = *(const v16h*)(bp + (size_t) 0 * N_NODES);
    v16h b1 = *(const v16h*)(bp + (size_t)16 * N_NODES);
    v16h b2 = *(const v16h*)(bp + (size_t)32 * N_NODES);
    v16h b3 = *(const v16h*)(bp + (size_t)48 * N_NODES);

    d0 = __builtin_amdgcn_wmma_f32_16x16x32_f16(false, p, false, b0,    (short)0, d0, false, false);
    d1 = __builtin_amdgcn_wmma_f32_16x16x32_f16(false, p, false, b1,    (short)0, d1, false, false);
    d2 = __builtin_amdgcn_wmma_f32_16x16x32_f16(false, p, false, b2,    (short)0, d2, false, false);
    d3 = __builtin_amdgcn_wmma_f32_16x16x32_f16(false, p, false, b3,    (short)0, d3, false, false);
    d4 = __builtin_amdgcn_wmma_f32_16x16x32_f16(false, p, false, bones, (short)0, d4, false, false);
  }

  // d4[r] = full row sum of P over this slice (WMMA summed all K groups);
  // every lane in a half holds the same value, so 2 lanes publish 8 rows each.
  const int rhalf = (lane >> 4) * 8;
  if (m == 0) {
#pragma unroll
    for (int r = 0; r < 8; ++r) atomicAdd(&lbuf[r0 + r + rhalf], d4[r]);
  }

#pragma unroll
  for (int r = 0; r < 8; ++r) {
    float* ap = acc + (size_t)(r0 + r + rhalf) * OUT_DIM + m;
    atomicAdd(ap + 0,  d0[r]);
    atomicAdd(ap + 16, d1[r]);
    atomicAdd(ap + 32, d2[r]);
    atomicAdd(ap + 48, d3[r]);
  }
}

// ---------------------------------------------------------------------------
// Phase 3: out = elu(acc / l)
// ---------------------------------------------------------------------------
__global__ void k_fin(const float* __restrict__ acc, const float* __restrict__ lbuf,
                      float* __restrict__ out) {
  int idx = blockIdx.x * blockDim.x + threadIdx.x;
  int i = idx >> 6;
  float l = lbuf[i];
  l = (l > 0.f) ? l : 1.0f;
  float x = acc[idx] / l;
  out[idx] = (x > 0.f) ? x : (__builtin_amdgcn_exp2f(x * LOG2E) - 1.0f);
}

// ---------------------------------------------------------------------------
extern "C" void kernel_launch(void* const* d_in, const int* in_sizes, int n_in,
                              void* d_out, int out_size, void* d_ws, size_t ws_size,
                              hipStream_t stream) {
  const float* h = (const float*)d_in[0];
  const int*   A = (const int*)d_in[1];
  const float* W = (const float*)d_in[2];
  const float* a = (const float*)d_in[3];
  float* out = (float*)d_out;

  // workspace layout
  float*     Whf  = (float*)d_ws;                               // 8192*64 f32
  _Float16*  WhT  = (_Float16*)(Whf + N_NODES * OUT_DIM);       // 64*8192 f16
  float*     f1   = (float*)(WhT + OUT_DIM * N_NODES);          // 8192 f32
  float*     f2   = f1 + N_NODES;                               // 8192 f32
  float*     f2m  = f2 + N_NODES;                               // 1 f32 (padded)
  float*     acc  = f2m + 16;                                   // 8192*64 f32
  float*     lbuf = acc + N_NODES * OUT_DIM;                    // 8192 f32

  hipMemsetAsync(acc, 0, (size_t)(N_NODES * OUT_DIM + N_NODES) * sizeof(float), stream);

  k_wh  <<<64, 256, 0, stream>>>(h, W, Whf, WhT);
  k_f12 <<<N_NODES / 256, 256, 0, stream>>>(Whf, a, f1, f2);
  k_f2max<<<1, 256, 0, stream>>>(f2, f2m);
  k_attn<<<(512 * NSLICE * 32) / 256, 256, 0, stream>>>(A, WhT, f1, f2, f2m, acc, lbuf);
  k_fin <<<(N_NODES * OUT_DIM) / 256, 256, 0, stream>>>(acc, lbuf, out);
}
